// SSGC_50405736186129
// MI455X (gfx1250) — matrix operand order repaired and checked by reading, not compiled, not run on hardware
//
#include <hip/hip_runtime.h>

// ---------------------------------------------------------------------------
// SSGConv (alpha=0, K=8) for MI455X / gfx1250.
//   Phase 1: degree + symmetric-norm edge weights (atomics).
//   Phase 2: 8 propagation hops, ping-pong buffers, acc folded into hop init.
//   Phase 3: h_out -> bf16, GEMM1 [N,512]x[512,256] via v_wmma_f32_16x16x32_bf16,
//            16x64 tile per wave (A fragment reused across 4 WMMAs).
//   Phase 4: GEMM2 [N,256]x[256,16(pad of 7)] via WMMA; B held in VGPRs and
//            reused across 4 M-tiles per wave; f32 out + bias.
// ---------------------------------------------------------------------------

#define F_IN  512
#define HID   256
#define CLS   7
#define KHOPS 8

typedef __attribute__((ext_vector_type(8)))  __bf16 v8bf;
typedef __attribute__((ext_vector_type(16))) __bf16 v16bf;
typedef __attribute__((ext_vector_type(8)))  float  v8f;

static __device__ __forceinline__ unsigned short f32_bf16(float f) {
  unsigned int u = __float_as_uint(f);
  u += 0x7FFFu + ((u >> 16) & 1u);   // round-to-nearest-even
  return (unsigned short)(u >> 16);
}

// ---------------- Phase 1: degrees & edge weights ----------------

__global__ void k_deg_init(float* deg, int n) {
  int i = blockIdx.x * blockDim.x + threadIdx.x;
  if (i < n) deg[i] = 1.0f;                       // self-loop
}

__global__ void k_deg_count(const int* __restrict__ dst, float* deg, int e) {
  int i = blockIdx.x * blockDim.x + threadIdx.x;
  if (i < e) atomicAdd(&deg[dst[i]], 1.0f);
}

__global__ void k_dinv(float* d, int n) {
  int i = blockIdx.x * blockDim.x + threadIdx.x;
  if (i < n) d[i] = rsqrtf(d[i]);                 // deg >= 1 always
}

__global__ void k_edge_w(const int* __restrict__ src, const int* __restrict__ dst,
                         const float* __restrict__ dinv, float* __restrict__ w, int e) {
  int i = blockIdx.x * blockDim.x + threadIdx.x;
  if (i < e) w[i] = dinv[src[i]] * dinv[dst[i]];
}

// ---------------- Phase 2: propagation ----------------

// h_next = selfweight * h_cur ; acc op on h_cur (mode 0: none, 1: acc=h, 2: acc+=h)
__global__ void k_hop_init(const float* __restrict__ hc, float* __restrict__ hn,
                           const float* __restrict__ dinv, float* __restrict__ acc,
                           int mode, int n) {
  long gid = (long)blockIdx.x * blockDim.x + threadIdx.x;   // over n * (F_IN/4)
  if (gid >= (long)n * (F_IN / 4)) return;
  int node = (int)(gid >> 7);                               // F_IN/4 == 128
  float di = dinv[node];
  float sw = di * di;
  float4 v = ((const float4*)hc)[gid];
  float4 o = make_float4(v.x * sw, v.y * sw, v.z * sw, v.w * sw);
  ((float4*)hn)[gid] = o;
  if (mode == 1) {
    ((float4*)acc)[gid] = v;
  } else if (mode == 2) {
    float4 a = ((float4*)acc)[gid];
    a.x += v.x; a.y += v.y; a.z += v.z; a.w += v.w;
    ((float4*)acc)[gid] = a;
  }
}

// one block per edge; 128 threads, 4 features each (coalesced float4 gather,
// f32 atomic scatter)
__global__ __launch_bounds__(128)
void k_scatter(const float* __restrict__ h, float* __restrict__ hn,
               const int* __restrict__ src, const int* __restrict__ dst,
               const float* __restrict__ w) {
  int e = blockIdx.x;
  int s = src[e];
  int d = dst[e];
  float we = w[e];
  int t = threadIdx.x;
  float4 v = ((const float4*)(h + (size_t)s * F_IN))[t];
  float* o = hn + (size_t)d * F_IN + 4 * t;
  atomicAdd(o + 0, v.x * we);
  atomicAdd(o + 1, v.y * we);
  atomicAdd(o + 2, v.z * we);
  atomicAdd(o + 3, v.w * we);
}

// h_out_bf16 = (acc + h_last) / K
__global__ void k_finalize(const float* __restrict__ acc, const float* __restrict__ hl,
                           unsigned short* __restrict__ outbf, long n4) {
  long gid = (long)blockIdx.x * blockDim.x + threadIdx.x;
  if (gid >= n4) return;
  float4 a = ((const float4*)acc)[gid];
  float4 h = ((const float4*)hl)[gid];
  const float s = 1.0f / (float)KHOPS;
  ushort4 o;
  o.x = f32_bf16((a.x + h.x) * s);
  o.y = f32_bf16((a.y + h.y) * s);
  o.z = f32_bf16((a.z + h.z) * s);
  o.w = f32_bf16((a.w + h.w) * s);
  ((ushort4*)outbf)[gid] = o;
}

// ---------------- weight conversion ----------------

__global__ void k_cvt_bf16(const float* __restrict__ in, unsigned short* __restrict__ out, int n) {
  int i = blockIdx.x * blockDim.x + threadIdx.x;
  if (i < n) out[i] = f32_bf16(in[i]);
}

// lin2_w [CLS,HID] -> padded [16,HID] bf16, rows >= CLS zeroed
__global__ void k_cvt_w2pad(const float* __restrict__ in, unsigned short* __restrict__ out) {
  int i = blockIdx.x * blockDim.x + threadIdx.x;
  if (i >= 16 * HID) return;
  int o = i / HID, k = i % HID;
  out[i] = (o < CLS) ? f32_bf16(in[o * HID + k]) : (unsigned short)0;
}

// ---------------- Phase 3/4: WMMA GEMMs ----------------
// A VGPR layout (16-bit, 16x32): lane L holds row M=L%16;
//   elems 0..15 -> K = ks + (L/16)*16 + j (contiguous order used here, same as B)
// B VGPR layout (16-bit, 32x16): lane L holds col N=L%16;
//   elems 0..15 -> K = ks + (L/16)*16 + j
// C/D f32: lane L, vgpr r -> (M = r + 8*(L/16), N = L%16)

static __device__ __forceinline__ v16bf load_afrag(const __bf16* arow, int ks) {
  v8bf a0 = *(const v8bf*)(arow + ks);
  v8bf a1 = *(const v8bf*)(arow + ks + 8);
  return __builtin_shufflevector(a0, a1, 0,1,2,3,4,5,6,7,8,9,10,11,12,13,14,15);
}

static __device__ __forceinline__ v16bf load_bfrag(const __bf16* bcol, int ks) {
  v8bf b0 = *(const v8bf*)(bcol + ks);
  v8bf b1 = *(const v8bf*)(bcol + ks + 8);
  return __builtin_shufflevector(b0, b1, 0,1,2,3,4,5,6,7,8,9,10,11,12,13,14,15);
}

// 16x64 output tile per wave: one A fragment feeds 4 WMMAs per K-step.
__global__ __launch_bounds__(32)
void k_gemm1(const unsigned short* __restrict__ A,   // [N, F_IN] bf16
             const unsigned short* __restrict__ B,   // [HID, F_IN] bf16 (B[k,o] = lin1_w[o,k])
             const float* __restrict__ bias,         // [HID]
             unsigned short* __restrict__ Cout,      // [N, HID] bf16
             int nrows) {
  const int mt = blockIdx.x, ntg = blockIdx.y;       // ntg covers 64 output cols
  const int lane = threadIdx.x;
  const int l16 = lane & 15, half = lane >> 4;
  int row = mt * 16 + l16;
  if (row >= nrows) row = nrows - 1;                 // N%16==0 in practice

  const __bf16* arow = (const __bf16*)(A + (size_t)row * F_IN) + half * 16;
  const __bf16* bcol[4];
#pragma unroll
  for (int j = 0; j < 4; ++j) {
    const int col = ntg * 64 + j * 16 + l16;
    bcol[j] = (const __bf16*)(B + (size_t)col * F_IN) + half * 16;
  }

  v8f c[4];
#pragma unroll
  for (int j = 0; j < 4; ++j) c[j] = (v8f){0.f,0.f,0.f,0.f,0.f,0.f,0.f,0.f};

#pragma unroll 4
  for (int ks = 0; ks < F_IN; ks += 32) {
    v16bf a = load_afrag(arow, ks);
#pragma unroll
    for (int j = 0; j < 4; ++j) {
      v16bf b = load_bfrag(bcol[j], ks);
      c[j] = __builtin_amdgcn_wmma_f32_16x16x32_bf16(false, a, false, b, (short)0, c[j],
                                                     false, false);
    }
  }

#pragma unroll
  for (int j = 0; j < 4; ++j) {
    const int col = ntg * 64 + j * 16 + l16;
    const float bv = bias[col];
#pragma unroll
    for (int r = 0; r < 8; ++r) {
      const int m = mt * 16 + r + half * 8;
      if (m < nrows) Cout[(size_t)m * HID + col] = f32_bf16(c[j][r] + bv);
    }
  }
}

// B (256x16, padded cols of lin2_w^T) is preloaded into 8 v16bf fragments and
// reused across MT2 consecutive M-tiles per wave.
#define MT2 4
__global__ __launch_bounds__(32)
void k_gemm2(const unsigned short* __restrict__ A,   // [N, HID] bf16
             const unsigned short* __restrict__ B,   // [16, HID] bf16, padded
             const float* __restrict__ bias,         // [CLS]
             float* __restrict__ out,                // [N, CLS]
             int nrows) {
  const int lane = threadIdx.x;
  const int l16 = lane & 15, half = lane >> 4;
  const int ntiles = (nrows + 15) / 16;

  const __bf16* bcol = (const __bf16*)(B + (size_t)l16 * HID) + half * 16;
  v16bf bf[HID / 32];
#pragma unroll
  for (int kk = 0; kk < HID / 32; ++kk) bf[kk] = load_bfrag(bcol, kk * 32);

  for (int t = 0; t < MT2; ++t) {
    const int mt = blockIdx.x * MT2 + t;
    if (mt >= ntiles) break;                         // uniform across the wave
    int row = mt * 16 + l16;
    if (row >= nrows) row = nrows - 1;
    const __bf16* arow = (const __bf16*)(A + (size_t)row * HID) + half * 16;

    v8f c = {0.f, 0.f, 0.f, 0.f, 0.f, 0.f, 0.f, 0.f};
#pragma unroll
    for (int kk = 0; kk < HID / 32; ++kk) {
      v16bf a = load_afrag(arow, kk * 32);
      c = __builtin_amdgcn_wmma_f32_16x16x32_bf16(false, a, false, bf[kk], (short)0, c,
                                                  false, false);
    }

    if (l16 < CLS) {
      const float bv = bias[l16];
#pragma unroll
      for (int r = 0; r < 8; ++r) {
        const int m = mt * 16 + r + half * 8;
        if (m < nrows) out[(size_t)m * CLS + l16] = c[r] + bv;
      }
    }
  }
}

// ---------------------------------------------------------------------------

extern "C" void kernel_launch(void* const* d_in, const int* in_sizes, int n_in,
                              void* d_out, int out_size, void* d_ws, size_t ws_size,
                              hipStream_t stream) {
  const float* x   = (const float*)d_in[0];
  const int*  eidx = (const int*)d_in[1];
  const float* w1  = (const float*)d_in[2];
  const float* b1  = (const float*)d_in[3];
  const float* w2  = (const float*)d_in[4];
  const float* b2  = (const float*)d_in[5];
  float* out = (float*)d_out;

  const int N = in_sizes[0] / F_IN;
  const int E = in_sizes[1] / 2;
  const int* src = eidx;
  const int* dst = eidx + E;

  // workspace carve (~781 MB total)
  char* p = (char*)d_ws;
  auto take = [&](size_t bytes) -> char* {
    char* r = p;
    p += (bytes + 255) & ~(size_t)255;
    return r;
  };
  float* dinv = (float*)take((size_t)N * 4);
  float* ew   = (float*)take((size_t)E * 4);
  float* hA   = (float*)take((size_t)N * F_IN * 4);
  float* hB   = (float*)take((size_t)N * F_IN * 4);
  float* acc  = (float*)take((size_t)N * F_IN * 4);
  unsigned short* houtbf = (unsigned short*)take((size_t)N * F_IN * 2);
  unsigned short* w1bf   = (unsigned short*)take((size_t)HID * F_IN * 2);
  unsigned short* w2bf   = (unsigned short*)take((size_t)16 * HID * 2);
  unsigned short* hidbf  = (unsigned short*)take((size_t)N * HID * 2);
  (void)ws_size; (void)n_in; (void)out_size;

  const int TB = 256;
  // Phase 1
  k_deg_init<<<(N + TB - 1) / TB, TB, 0, stream>>>(dinv, N);
  k_deg_count<<<(E + TB - 1) / TB, TB, 0, stream>>>(dst, dinv, E);
  k_dinv<<<(N + TB - 1) / TB, TB, 0, stream>>>(dinv, N);
  k_edge_w<<<(E + TB - 1) / TB, TB, 0, stream>>>(src, dst, dinv, ew, E);

  // weight conversion (independent of propagation)
  k_cvt_bf16<<<(HID * F_IN + TB - 1) / TB, TB, 0, stream>>>(w1, w1bf, HID * F_IN);
  k_cvt_w2pad<<<(16 * HID + TB - 1) / TB, TB, 0, stream>>>(w2, w2bf);

  // Phase 2: 8 hops
  const long n4 = (long)N * (F_IN / 4);
  const int eg = (int)((n4 + TB - 1) / TB);
  const float* cur = x;
  float* nxt = hA;
  for (int k = 1; k <= KHOPS; ++k) {
    int mode = (k == 1) ? 0 : (k == 2 ? 1 : 2);
    k_hop_init<<<eg, TB, 0, stream>>>(cur, nxt, dinv, acc, mode, N);
    k_scatter<<<E, 128, 0, stream>>>(cur, nxt, src, dst, ew);
    cur = nxt;
    nxt = (nxt == hA) ? hB : hA;
  }
  // cur == h_K ; acc == h_1 + ... + h_{K-1}
  k_finalize<<<eg, TB, 0, stream>>>(acc, cur, houtbf, n4);

  // Phase 3: GEMM1 (WMMA bf16 -> f32), 16x64 tile per wave
  dim3 g1((N + 15) / 16, HID / 64);
  k_gemm1<<<g1, 32, 0, stream>>>(houtbf, w1bf, b1, hidbf, N);

  // Phase 4: GEMM2 (WMMA bf16 -> f32) + bias, f32 output
  const int ntiles = (N + 15) / 16;
  k_gemm2<<<(ntiles + MT2 - 1) / MT2, 32, 0, stream>>>(hidbf, w2bf, b2, out, N);
}
